// ContentAwareBlend_30099130810967
// MI455X (gfx1250) — matrix operand order repaired and checked by reading, not compiled, not run on hardware
//
#include <hip/hip_runtime.h>
#include <hip/hip_bf16.h>
#include <stddef.h>

#define B_   8
#define C_   256
#define H_   128
#define W_   128
#define HW_  (H_*W_)
#define KTOT 2304           // 9 taps * 256 input channels (tap-major K)
#define HP_  130            // padded spatial (pad=1)

typedef __attribute__((ext_vector_type(16))) __bf16 v16bf;
typedef __attribute__((ext_vector_type(8)))  __bf16 v8bf;
typedef __attribute__((ext_vector_type(8)))  float  v8f;

static __device__ __forceinline__ __bf16 f2bf(float f) {
  unsigned u = __float_as_uint(f);
  unsigned r = (u + 0x7FFFu + ((u >> 16) & 1u)) >> 16;   // round-to-nearest-even
  unsigned short s = (unsigned short)r;
  return __builtin_bit_cast(__bf16, s);
}

// ---------------------------------------------------------------------------
// x [b][c][y][x] f32 (NCHW)  ->  xh [b][y+1][x+1][c] bf16 (padded NHWC, zeros)
// One block per (row, x-half, b); transpose through LDS for coalesced writes.
// ---------------------------------------------------------------------------
__global__ __launch_bounds__(256) void make_xh(const float* __restrict__ x,
                                               __bf16* __restrict__ xh) {
  const int yy1   = blockIdx.x;   // 0..129 (padded row)
  const int xhalf = blockIdx.y;   // 0..1
  const int b     = blockIdx.z;
  const int t     = threadIdx.x;
  __bf16* row = xh + ((size_t)b * HP_ + yy1) * HP_ * 256;
  const uint4 zz = {0u, 0u, 0u, 0u};
  if (yy1 == 0 || yy1 == HP_ - 1) {        // fully zero padded rows
    if (xhalf == 0) {
      for (int i = t * 8; i < HP_ * 256; i += 256 * 8)
        *(uint4*)(row + i) = zz;
    }
    return;
  }
  __shared__ __align__(16) __bf16 Ls[64][264];
  const int y = yy1 - 1;
  {  // load: one channel per thread, 64 pixels of this half
    const float* src = x + (((size_t)b * C_ + t) * H_ + y) * W_ + xhalf * 64;
    for (int xx = 0; xx < 64; ++xx) Ls[xx][t] = f2bf(src[xx]);
  }
  __syncthreads();
  if (t < 32) {  // zero the x-borders (padded pixels 0 and 129)
    if (xhalf == 0) *(uint4*)(row + 0 * 256 + t * 8) = zz;
    else            *(uint4*)(row + (HP_ - 1) * 256 + t * 8) = zz;
  }
  {  // coalesced NHWC writes: pixel p = t>>2, 64-channel slice (t&3)
    const int xx = t >> 2;
    const int cb = (t & 3) * 64;
    __bf16* dst = row + (size_t)(xhalf * 64 + xx + 1) * 256 + cb;
#pragma unroll
    for (int i = 0; i < 8; ++i)
      *(uint4*)(dst + i * 8) = *(const uint4*)&Ls[xx][cb + i * 8];
  }
}

// bs_w [n=256][cin=256][3][3] -> wTt [n=256][k=2304], k = tap*256 + cin
__global__ __launch_bounds__(256) void prep_w_conv(const float* __restrict__ w,
                                                   __bf16* __restrict__ o) {
  int idx = blockIdx.x * 256 + threadIdx.x;   // 589824 total
  int n  = idx / KTOT;
  int k  = idx % KTOT;
  int tp = k >> 8;
  int ci = k & 255;
  o[idx] = f2bf(w[((size_t)n * 256 + ci) * 9 + tp]);
}

// ke_w2 [25][256] -> w2t [32][256] bf16, zero padded rows 25..31
__global__ __launch_bounds__(256) void prep_w2(const float* __restrict__ w,
                                               __bf16* __restrict__ o) {
  int idx = blockIdx.x * 256 + threadIdx.x;   // 8192 total
  int n = idx >> 8, k = idx & 255;
  o[idx] = f2bf(n < 25 ? w[n * 256 + k] : 0.0f);
}

// ---------------------------------------------------------------------------
// Dense 3x3 conv + BN + SiLU, implicit GEMM, v_wmma_f32_16x16x32_bf16.
// A-fragments load DIRECTLY from padded NHWC global: no branches, no LDS,
// no barriers inside the K loop. M-block = image row (128 px), N-block = 64.
// ---------------------------------------------------------------------------
__global__ __launch_bounds__(256) void conv3x3_bn_silu_wmma(
    const __bf16* __restrict__ xh, const __bf16* __restrict__ wTt,
    const float* __restrict__ bias, const float* __restrict__ bng,
    const float* __restrict__ bnb,  const float* __restrict__ bnm,
    const float* __restrict__ bnv,  float* __restrict__ out) {
  __shared__ float Cs[64][129];                  // [n][m] epilogue transpose
  const int nb   = blockIdx.x;                   // 0..3 (64-channel block)
  const int by   = blockIdx.y;                   // 0..1023
  const int b    = by >> 7, y = by & 127;
  const int t    = threadIdx.x;
  const int lane = t & 31;
  const int wave = t >> 5;                       // 8 waves: 4 (M) x 2 (N)
  const int wm   = wave >> 1, wn = wave & 1;
  const int lrow = lane & 15;
  const int koffA = (lane & 16) ? 16 : 0;        // contiguous k order: k = 16*(lane/16) + s
  const int koffB = (lane & 16) ? 16 : 0;        // contiguous k order, same as A

  v8f acc00 = {}, acc01 = {}, acc10 = {}, acc11 = {};

  const int px0 = wm * 32 + lrow;                // pixel of frag0 (frag1: +16)
  const __bf16* pb0 =
      wTt + (size_t)(nb * 64 + wn * 32 + lrow) * KTOT + koffB;
  const __bf16* pb1 = pb0 + (size_t)16 * KTOT;

  for (int tap = 0; tap < 9; ++tap) {
    const int dy = tap / 3 - 1, dx = tap % 3 - 1;
    const __bf16* pa0 =
        xh + (((size_t)b * HP_ + (y + dy + 1)) * HP_ + (px0 + dx + 1)) * 256 + koffA;
    const __bf16* pa1 = pa0 + (size_t)16 * 256;
    const int kb0 = tap * 256;
#pragma unroll 2
    for (int cc = 0; cc < 8; ++cc) {
      const int co = cc * 32;
      v8bf lo, hi;
      lo = *(const v8bf*)(pa0 + co);
      hi = *(const v8bf*)(pa0 + co + 8);
      const v16bf a0 = __builtin_shufflevector(lo, hi, 0,1,2,3,4,5,6,7,8,9,10,11,12,13,14,15);
      lo = *(const v8bf*)(pa1 + co);
      hi = *(const v8bf*)(pa1 + co + 8);
      const v16bf a1 = __builtin_shufflevector(lo, hi, 0,1,2,3,4,5,6,7,8,9,10,11,12,13,14,15);
      const v16bf bf0 = *(const v16bf*)(pb0 + kb0 + co);
      const v16bf bf1 = *(const v16bf*)(pb1 + kb0 + co);
      acc00 = __builtin_amdgcn_wmma_f32_16x16x32_bf16(false, a0, false, bf0, (short)0, acc00, false, false);
      acc01 = __builtin_amdgcn_wmma_f32_16x16x32_bf16(false, a0, false, bf1, (short)0, acc01, false, false);
      acc10 = __builtin_amdgcn_wmma_f32_16x16x32_bf16(false, a1, false, bf0, (short)0, acc10, false, false);
      acc11 = __builtin_amdgcn_wmma_f32_16x16x32_bf16(false, a1, false, bf1, (short)0, acc11, false, false);
    }
  }
  {  // spill C (transposed) to LDS for coalesced stores
    const int mhalf = (lane & 16) ? 8 : 0;
    const int nlo = wn * 32 + lrow;
    const int m0 = wm * 32 + mhalf;
#pragma unroll
    for (int v = 0; v < 8; ++v) {
      Cs[nlo     ][m0 + v]      = acc00[v];
      Cs[nlo + 16][m0 + v]      = acc01[v];
      Cs[nlo     ][m0 + 16 + v] = acc10[v];
      Cs[nlo + 16][m0 + 16 + v] = acc11[v];
    }
  }
  __syncthreads();
  {  // fused BN + SiLU epilogue, channel-row coalesced stores
    const int x  = t & 127;
    const int nh = t >> 7;
#pragma unroll 4
    for (int i = 0; i < 32; ++i) {
      const int n  = i * 2 + nh;
      const int ch = nb * 64 + n;
      const float inv = rsqrtf(bnv[ch] + 1e-5f);
      const float sc  = bng[ch] * inv;
      const float sh  = bnb[ch] + (bias[ch] - bnm[ch]) * sc;
      const float vv  = Cs[n][x] * sc + sh;
      const float s   = vv / (1.0f + __expf(-vv));
      out[(((size_t)b * C_ + ch) * H_ + y) * W_ + x] = s;
    }
  }
}

// ---------------------------------------------------------------------------
// Grouped 3x3 conv + BN + SiLU -> h in NHWC bf16 [b][y][x][256]
// ---------------------------------------------------------------------------
__global__ __launch_bounds__(256) void kenc_conv(
    const float* __restrict__ x, const float* __restrict__ w1,
    const float* __restrict__ b1, const float* __restrict__ bng,
    const float* __restrict__ bnb, const float* __restrict__ bnm,
    const float* __restrict__ bnv, __bf16* __restrict__ hn) {
  __shared__ float xs[8][3][130];
  __shared__ float ws[8][8][9];
  __shared__ __align__(16) __bf16 hs[128][8];
  const int y = blockIdx.x, g = blockIdx.y, b = blockIdx.z;
  const int t = threadIdx.x;
  for (int idx = t; idx < 8 * 3 * 130; idx += 256) {
    int ci  = idx / 390;
    int rem = idx % 390;
    int r   = rem / 130;
    int xx  = rem % 130 - 1;
    int yy  = y + r - 1;
    float v = 0.0f;
    if (yy >= 0 && yy < H_ && xx >= 0 && xx < W_)
      v = x[(((size_t)b * C_ + g * 8 + ci) * H_ + yy) * W_ + xx];
    xs[ci][r][rem % 130] = v;
  }
  for (int idx = t; idx < 576; idx += 256)
    (&ws[0][0][0])[idx] = w1[(size_t)g * 576 + idx];
  __syncthreads();
  const int oc = t >> 5, lane = t & 31;
  const int ch = g * 8 + oc;
  const float bb  = b1[ch];
  const float inv = rsqrtf(bnv[ch] + 1e-5f);
  const float sc  = bng[ch] * inv;
  const float sh  = bnb[ch] - bnm[ch] * sc;
#pragma unroll
  for (int xi = 0; xi < 4; ++xi) {
    const int xx = lane + xi * 32;
    float acc = bb;
#pragma unroll
    for (int ci = 0; ci < 8; ++ci)
#pragma unroll
      for (int ky = 0; ky < 3; ++ky)
#pragma unroll
        for (int kx = 0; kx < 3; ++kx)
          acc += xs[ci][ky][xx + kx] * ws[oc][ci][ky * 3 + kx];
    const float vv = acc * sc + sh;
    const float s  = vv / (1.0f + __expf(-vv));
    hs[xx][oc] = f2bf(s);
  }
  __syncthreads();
  if (t < 128)   // 16B per pixel into NHWC slot of this group
    *(uint4*)(hn + ((size_t)(b * H_ + y) * W_ + t) * 256 + g * 8) =
        *(const uint4*)&hs[t][0];
}

// ---------------------------------------------------------------------------
// 1x1 conv (256 -> 25 pad 32) via WMMA from NHWC h; zero LDS, zero branches.
// ---------------------------------------------------------------------------
__global__ __launch_bounds__(256) void conv1x1_logits_wmma(
    const __bf16* __restrict__ hn, const __bf16* __restrict__ w2t,
    const float* __restrict__ b2, float* __restrict__ logit) {
  const int y = blockIdx.x, b = blockIdx.y;
  const int t = threadIdx.x;
  const int lane = t & 31, wave = t >> 5;      // 8 waves, each 16 pixels (M)
  const int lrow = lane & 15;
  const int koffA = (lane & 16) ? 16 : 0;
  const int koffB = (lane & 16) ? 16 : 0;
  v8f acc0 = {}, acc1 = {};
  const __bf16* pa =
      hn + ((size_t)(b * H_ + y) * W_ + (wave * 16 + lrow)) * 256 + koffA;
  const __bf16* pb0 = w2t + (size_t)lrow * 256 + koffB;
  const __bf16* pb1 = pb0 + 16 * 256;
#pragma unroll
  for (int cc = 0; cc < 8; ++cc) {
    const int co = cc * 32;
    const v8bf lo = *(const v8bf*)(pa + co);
    const v8bf hi = *(const v8bf*)(pa + co + 8);
    const v16bf a = __builtin_shufflevector(lo, hi, 0,1,2,3,4,5,6,7,8,9,10,11,12,13,14,15);
    const v16bf bf0 = *(const v16bf*)(pb0 + co);
    const v16bf bf1 = *(const v16bf*)(pb1 + co);
    acc0 = __builtin_amdgcn_wmma_f32_16x16x32_bf16(false, a, false, bf0, (short)0, acc0, false, false);
    acc1 = __builtin_amdgcn_wmma_f32_16x16x32_bf16(false, a, false, bf1, (short)0, acc1, false, false);
  }
  const int mhalf = (lane & 16) ? 8 : 0;
#pragma unroll
  for (int v = 0; v < 8; ++v) {
    const int m = wave * 16 + mhalf + v;
    const size_t base = (((size_t)(b * H_ + y)) * W_ + m) * 32;
    const int n0 = lrow, n1 = lrow + 16;
    float v0 = acc0[v] + b2[n0];              // n0 < 16 < 25 always
    float v1 = acc1[v];
    if (n1 < 25) v1 += b2[n1];
    logit[base + n0] = v0;
    logit[base + n1] = v1;
  }
}

// ---------------------------------------------------------------------------
// Gate: g[b] = sigmoid( (1/HW) * sum_{c,px} x * g_w[c] + g_b )
// ---------------------------------------------------------------------------
__global__ __launch_bounds__(256) void gate_kernel(
    const float* __restrict__ x, const float* __restrict__ gw,
    const float* __restrict__ gb, float* __restrict__ gout) {
  __shared__ float red[256];
  const int b = blockIdx.x, t = threadIdx.x;
  float a = 0.0f;
  for (int c = 0; c < C_; ++c) {
    const float wv = gw[c];
    const float* p = x + ((size_t)b * C_ + c) * HW_;
    for (int i = t; i < HW_; i += 256) a += p[i] * wv;
  }
  red[t] = a;
  __syncthreads();
  for (int s = 128; s > 0; s >>= 1) {
    if (t < s) red[t] += red[t + s];
    __syncthreads();
  }
  if (t == 0) {
    const float m = red[0] / (float)HW_ + gb[0];
    gout[b] = 1.0f / (1.0f + __expf(-m));
  }
}

// ---------------------------------------------------------------------------
// CARAFE (softmax over 25 taps, reflect pad 2) + final blend, in-place on out
// ---------------------------------------------------------------------------
__global__ __launch_bounds__(256) void carafe_blend(
    const float* __restrict__ x, const float* __restrict__ logit,
    const float* __restrict__ gout, float* __restrict__ out) {
  __shared__ float wgt[25][132];
  __shared__ float xs[16][5][132];
  const int y = blockIdx.x, b = blockIdx.y;
  const int t = threadIdx.x;
  if (t < 128) {  // per-pixel softmax of 25 logits
    const float* lp = logit + (((size_t)(b * H_ + y)) * W_ + t) * 32;
    float l[25], mx = -1e30f;
#pragma unroll
    for (int n = 0; n < 25; ++n) { l[n] = lp[n]; mx = fmaxf(mx, l[n]); }
    float s = 0.0f;
#pragma unroll
    for (int n = 0; n < 25; ++n) { l[n] = __expf(l[n] - mx); s += l[n]; }
    const float r = 1.0f / s;
#pragma unroll
    for (int n = 0; n < 25; ++n) wgt[n][t] = l[n] * r;
  }
  const float g  = gout[b];
  const float gm = 1.0f - g;
  const int ci = t >> 4;
  const int xseg = (t & 15) * 8;
  for (int chunk = 0; chunk < 16; ++chunk) {
    __syncthreads();
    for (int idx = t; idx < 16 * 5 * 132; idx += 256) {
      const int c   = idx / 660;
      const int rem = idx % 660;
      const int rr  = rem / 132;
      const int xx  = rem % 132 - 2;
      const int yy  = y + rr - 2;
      const int ry  = yy < 0 ? -yy : (yy > 127 ? 254 - yy : yy);
      const int rx  = xx < 0 ? -xx : (xx > 127 ? 254 - xx : xx);
      xs[c][rr][rem % 132] =
          x[(((size_t)b * C_ + chunk * 16 + c) * H_ + ry) * W_ + rx];
    }
    __syncthreads();
    const int c = chunk * 16 + ci;
#pragma unroll
    for (int xi = 0; xi < 8; ++xi) {
      const int xx = xseg + xi;
      float acc = 0.0f;
#pragma unroll
      for (int n = 0; n < 25; ++n)
        acc += xs[ci][n / 5][xx + (n % 5)] * wgt[n][xx];
      const size_t o = (((size_t)b * C_ + c) * H_ + y) * W_ + xx;
      out[o] = g * acc + gm * out[o];   // out currently holds `static`
    }
  }
}

// ---------------------------------------------------------------------------
extern "C" void kernel_launch(void* const* d_in, const int* in_sizes, int n_in,
                              void* d_out, int out_size, void* d_ws, size_t ws_size,
                              hipStream_t stream) {
  const float* x      = (const float*)d_in[0];
  const float* ke_w1  = (const float*)d_in[1];
  const float* ke_b1  = (const float*)d_in[2];
  const float* ke_bng = (const float*)d_in[3];
  const float* ke_bnb = (const float*)d_in[4];
  const float* ke_bnm = (const float*)d_in[5];
  const float* ke_bnv = (const float*)d_in[6];
  const float* ke_w2  = (const float*)d_in[7];
  const float* ke_b2  = (const float*)d_in[8];
  const float* bs_w   = (const float*)d_in[9];
  const float* bs_b   = (const float*)d_in[10];
  const float* bs_bng = (const float*)d_in[11];
  const float* bs_bnb = (const float*)d_in[12];
  const float* bs_bnm = (const float*)d_in[13];
  const float* bs_bnv = (const float*)d_in[14];
  const float* g_w    = (const float*)d_in[15];
  const float* g_b    = (const float*)d_in[16];
  float* out = (float*)d_out;

  char* ws = (char*)d_ws;
  __bf16* xh   = (__bf16*)(ws);                  //  69,222,400 B (padded NHWC)
  __bf16* wTt  = (__bf16*)(ws + 69222400);       //   1,179,648 B
  __bf16* w2t  = (__bf16*)(ws + 70402048);       //      16,384 B
  __bf16* hn   = (__bf16*)(ws + 70418432);       //  67,108,864 B (NHWC h)
  float*  lgt  = (float*) (ws + 137527296);      //  16,777,216 B
  float*  gout = (float*) (ws + 154304512);      //          32 B (~147 MB)

  make_xh<<<dim3(HP_, 2, B_), 256, 0, stream>>>(x, xh);
  prep_w_conv<<<2304, 256, 0, stream>>>(bs_w, wTt);
  prep_w2<<<32, 256, 0, stream>>>(ke_w2, w2t);
  gate_kernel<<<8, 256, 0, stream>>>(x, g_w, g_b, gout);
  conv3x3_bn_silu_wmma<<<dim3(4, 1024), 256, 0, stream>>>(
      xh, wTt, bs_b, bs_bng, bs_bnb, bs_bnm, bs_bnv, out);
  kenc_conv<<<dim3(128, 32, 8), 256, 0, stream>>>(
      x, ke_w1, ke_b1, ke_bng, ke_bnb, ke_bnm, ke_bnv, hn);
  conv1x1_logits_wmma<<<dim3(128, 8), 256, 0, stream>>>(hn, w2t, ke_b2, lgt);
  carafe_blend<<<dim3(128, 8), 256, 0, stream>>>(x, lgt, gout, out);
}
